// DirectionalModel_83365315216145
// MI455X (gfx1250) — compile-verified, matrix operand order checked by reading, not run on hardware
//
#include <hip/hip_runtime.h>

// ---------------------------------------------------------------------------
// MI455X (gfx1250, wave32) implementation of the DirectionalModel reference.
// bf16 WMMA (v_wmma_f32_16x16x32_bf16) for all GEMMs, f32 accumulation.
// Weight tiles staged to LDS via async global->LDS when available.
// ---------------------------------------------------------------------------

typedef __bf16 bf16_t;
typedef __attribute__((ext_vector_type(16))) __bf16 v16bf;
typedef __attribute__((ext_vector_type(8)))  float  v8f;

// pointer types for the async global->LDS builtin (b128 = int4 payload)
typedef int v4i128 __attribute__((vector_size(16)));
typedef __attribute__((address_space(1))) v4i128* gas1_p128;
typedef __attribute__((address_space(3))) v4i128* gas3_p128;

#define BB   16
#define VV   1024
#define DDIM 16
#define EE   64
#define FFEA 16
#define HH   128
#define HDD  64
#define NBV  (BB * VV)          /* 16384 node rows   */
#define RR   (BB * VV * DDIM)   /* 262144 edge rows  */
#define KENC 96                 /* 80 padded to 96   */
#define KMAX 256                /* max GEMM K        */
#define BIGF 1e9f

// ---------------------------------------------------------------------------
// WMMA helpers (CDNA5 16x16x32 bf16, wave32)
// ---------------------------------------------------------------------------
__device__ __forceinline__ v8f vzero8() {
  v8f z = {0.f, 0.f, 0.f, 0.f, 0.f, 0.f, 0.f, 0.f};
  return z;
}

// A/B fragment loader. `base` is row-major [m, K] bf16 (for B, "rows" are
// output columns because weights are pre-transposed to [n*K + k]).
// ISA 16-bit A layout: lanes 0-15 -> M=lane, K = {0..7, 16..23};
// lanes 16-31 -> same M, K = {8..15, 24..31}.
__device__ __forceinline__ v16bf load_frag(const bf16_t* __restrict__ base, int stride,
                                           int mbase, int kbase, int lane) {
  int m    = mbase + (lane & 15);
  int half = (lane >> 4) & 1;
  const bf16_t* p = base + (long)m * stride + kbase + half * 8;
  v16bf f;
#pragma unroll
  for (int i = 0; i < 8; ++i) { f[i] = p[i]; f[i + 8] = p[16 + i]; }
  return f;
}

__device__ __forceinline__ v8f wmma_bf16(v16bf a, v16bf b, v8f c) {
  return __builtin_amdgcn_wmma_f32_16x16x32_bf16(false, a, false, b, (short)0, c,
                                                 false, false);
}

__device__ __forceinline__ float sigmoidf_(float x) { return 1.f / (1.f + __expf(-x)); }

// ---------------------------------------------------------------------------
// Generic GEMM: act(A[M,K]bf16 @ W[K,N] + bias) with W pre-swizzled to [n*K+k].
// The block's 64-column weight tile is staged to LDS once (async->LDS on
// gfx1250 when the builtin exists, else a vectorized VGPR bounce), then all
// 8 waves serve B fragments from LDS across every K step.
// Optional f32 store, bf16 store, and per-row atomic dot with dvec
// (fuses t . a_gat, tanh(.) @ W_dec2, tanh(.) @ W_dual2).
// grid = (M/128, N/64), block = 256 (8 waves, each owns a 16-row stripe).
// ---------------------------------------------------------------------------
__global__ __launch_bounds__(256) void gemm_bf16_kernel(
    const bf16_t* __restrict__ A, const bf16_t* __restrict__ Wt,
    const float* __restrict__ bias, float* __restrict__ Cf,
    bf16_t* __restrict__ Cbf, const float* __restrict__ dvec,
    float* __restrict__ dout, int N, int K, int act) {
  __shared__ bf16_t Blds[64 * KMAX];  // 32 KB max (K<=256)

  int wave = threadIdx.x >> 5, lane = threadIdx.x & 31;
  int half = lane >> 4;
  int rowbase = blockIdx.x * 128 + wave * 16;
  int colblk  = blockIdx.y * 64;

  // ---- stage the 64 x K weight tile (contiguous in pre-swizzled layout) ----
  {
    const char* gsrc = (const char*)(Wt + (size_t)colblk * K);
    char* ldst = (char*)Blds;
    int bytes = 64 * K * 2;
#if defined(__gfx1250__) && __has_builtin(__builtin_amdgcn_global_load_async_to_lds_b128)
    for (int i = threadIdx.x * 16; i < bytes; i += 256 * 16)
      __builtin_amdgcn_global_load_async_to_lds_b128(
          (gas1_p128)(void*)(gsrc + i),
          (gas3_p128)(void*)(ldst + i), 0, 0);
#if __has_builtin(__builtin_amdgcn_s_wait_asynccnt)
    __builtin_amdgcn_s_wait_asynccnt(0);
#endif
#else
    const uint4* g4 = (const uint4*)gsrc;
    uint4* l4 = (uint4*)ldst;
    for (int i = threadIdx.x; i < bytes / 16; i += 256) l4[i] = g4[i];
#endif
    __syncthreads();
  }

  v8f acc[4];
#pragma unroll
  for (int j = 0; j < 4; ++j) acc[j] = vzero8();

  const bf16_t* arow = A + (long)(rowbase + (lane & 15)) * K;
  for (int k = 0; k < K; k += 32) {
    v16bf a = load_frag(A, K, rowbase, k, lane);
    if (k + 32 < K) __builtin_prefetch(arow + k + 32, 0, 1);  // global_prefetch_b8
#pragma unroll
    for (int j = 0; j < 4; ++j) {
      v16bf b = load_frag(Blds, K, j * 16, k, lane);  // ds_load_b128 x2
      acc[j] = wmma_bf16(a, b, acc[j]);
    }
  }

#pragma unroll
  for (int j = 0; j < 4; ++j) {
    int col   = colblk + j * 16 + (lane & 15);
    float bia = bias ? bias[col] : 0.f;
    float dv  = dvec ? dvec[col] : 0.f;
#pragma unroll
    for (int r = 0; r < 8; ++r) {
      int row = rowbase + r + half * 8;
      float v = acc[j][r] + bia;
      if (act) v = tanhf(v);
      long idx = (long)row * N + col;
      if (Cf)  Cf[idx]  = v;
      if (Cbf) Cbf[idx] = (bf16_t)v;
      if (dout) {
        float p = v * dv;
#pragma unroll
        for (int o = 1; o < 16; o <<= 1) p += __shfl_xor(p, o, 16);
        if ((lane & 15) == 0) atomicAdd(&dout[row], p);
      }
    }
  }
}

// ---------------------------------------------------------------------------
// Fused GRU: for each 16-row tile compute gx = x@Wx (3 gate tiles) and
// gh = h@Wh (3 gate tiles) with 6 WMMA accumulators, then apply gate math.
// Weights (192 KB) stay in L2 (192 MB device L2 keeps them resident).
// Writes f32 h in place and new bf16 h to a separate buffer.
// grid = R/128, block = 256.
// ---------------------------------------------------------------------------
__global__ __launch_bounds__(256) void gru_kernel(
    const bf16_t* __restrict__ xbf, const bf16_t* __restrict__ hbf_in,
    const bf16_t* __restrict__ wx, const bf16_t* __restrict__ wh,
    const float* __restrict__ bgru, float* __restrict__ hf,
    bf16_t* __restrict__ hbf_out) {
  int wave = threadIdx.x >> 5, lane = threadIdx.x & 31;
  int half = lane >> 4;
  int rowbase = (blockIdx.x * 8 + wave) * 16;

  for (int j = 0; j < 8; ++j) {  // column tiles of H=128
    v8f accx[3], acch[3];
#pragma unroll
    for (int g = 0; g < 3; ++g) { accx[g] = vzero8(); acch[g] = vzero8(); }
    for (int k = 0; k < HH; k += 32) {
      v16bf a_x = load_frag(xbf,    HH, rowbase, k, lane);
      v16bf a_h = load_frag(hbf_in, HH, rowbase, k, lane);
#pragma unroll
      for (int g = 0; g < 3; ++g) {
        v16bf bx = load_frag(wx, HH, g * HH + j * 16, k, lane);
        accx[g] = wmma_bf16(a_x, bx, accx[g]);
        v16bf bh = load_frag(wh, HH, g * HH + j * 16, k, lane);
        acch[g] = wmma_bf16(a_h, bh, acch[g]);
      }
    }
    int col = j * 16 + (lane & 15);
#pragma unroll
    for (int r = 0; r < 8; ++r) {
      int row   = rowbase + r + half * 8;
      float z   = sigmoidf_(accx[0][r] + bgru[col]          + acch[0][r]);
      float rg  = sigmoidf_(accx[1][r] + bgru[HH + col]     + acch[1][r]);
      float cnd = tanhf    (accx[2][r] + bgru[2 * HH + col] + rg * acch[2][r]);
      long idx  = (long)row * HH + col;
      float hold = hf[idx];
      float hn   = z * hold + (1.f - z) * cnd;
      hf[idx]      = hn;
      hbf_out[idx] = (bf16_t)hn;
    }
  }
}

// ---------------------------------------------------------------------------
// Elementwise / reduction kernels
// ---------------------------------------------------------------------------
__global__ void zero_kernel(float* p, int n) {
  int t = blockIdx.x * blockDim.x + threadIdx.x;
  if (t < n) p[t] = 0.f;
}

// f32 [K,N] weight -> bf16 [n*Kp + k] (transposed, K zero-padded to Kp)
__global__ void packw_kernel(const float* __restrict__ src, bf16_t* __restrict__ dst,
                             int K, int Kp, int N) {
  int t = blockIdx.x * blockDim.x + threadIdx.x;
  if (t >= N * Kp) return;
  int n = t / Kp, k = t - n * Kp;
  dst[t] = (k < K) ? (bf16_t)src[(long)k * N + n] : (bf16_t)0.f;
}

// normalized embedding (max_norm = 1) -> bf16
__global__ void embn_kernel(const float* __restrict__ emb, bf16_t* __restrict__ out) {
  int v = blockIdx.x * blockDim.x + threadIdx.x;
  if (v >= VV) return;
  float s = 0.f;
  for (int k = 0; k < EE; ++k) { float e = emb[v * EE + k]; s += e * e; }
  float dn = fmaxf(sqrtf(s), 1.f);
  for (int k = 0; k < EE; ++k) out[v * EE + k] = (bf16_t)(emb[v * EE + k] / dn);
}

// encoder input rows: [emb_norm(64) | feat(16) | pad(16)]
__global__ void aenc_kernel(const bf16_t* __restrict__ embn,
                            const float* __restrict__ feat, bf16_t* __restrict__ a) {
  int t = blockIdx.x * blockDim.x + threadIdx.x;  // NBV * KENC
  int n = t / KENC, k = t - n * KENC;
  int v = n & (VV - 1);
  bf16_t val;
  if (k < EE)             val = embn[v * EE + k];
  else if (k < EE + FFEA) val = (bf16_t)feat[(long)n * FFEA + (k - EE)];
  else                    val = (bf16_t)0.f;
  a[t] = val;
}

// h0 = initial = gather(node_enc, adj) * valid  (f32 + bf16)
__global__ void hinit_kernel(const float* __restrict__ enc, const int* __restrict__ adj,
                             const int* __restrict__ nn, float* __restrict__ hf,
                             bf16_t* __restrict__ hbf) {
  int t = blockIdx.x * blockDim.x + threadIdx.x;  // RR * 32
  int row = t >> 5, c0 = (t & 31) * 4;
  int b = row / (VV * DDIM);
  int a = adj[row];
  float val = (a == nn[b]) ? 0.f : 1.f;
  int ac = a < VV ? (a < 0 ? 0 : a) : VV - 1;
  const float* sp = enc + ((long)b * VV + ac) * HH + c0;
  long o = (long)row * HH + c0;
#pragma unroll
  for (int c = 0; c < 4; ++c) {
    float v = val * sp[c];
    hf[o + c]  = v;
    hbf[o + c] = (bf16_t)v;
  }
}

// per-node masked softmax over the 16 slots + weighted message
// (one wave per node; shuffles lower to ds_bpermute/ds_swizzle)
__global__ void attn_kernel(const float* __restrict__ logits, const bf16_t* __restrict__ tbf,
                            const int* __restrict__ adj, const int* __restrict__ nn,
                            float* __restrict__ msg) {
  int wave = threadIdx.x >> 5, lane = threadIdx.x & 31;
  int node = blockIdx.x * 8 + wave;
  int b = node / VV;
  int rowbase = node * DDIM;
  int d = lane & 15;
  float m = (adj[rowbase + d] == nn[b]) ? 1.f : 0.f;
  float l = logits[rowbase + d] - BIGF * m;
  float mx = l;
#pragma unroll
  for (int o = 8; o; o >>= 1) mx = fmaxf(mx, __shfl_xor(mx, o, 16));
  float e = __expf(l - mx);
  float s = e;
#pragma unroll
  for (int o = 8; o; o >>= 1) s += __shfl_xor(s, o, 16);
  float attn = e / s;

  float acc[4] = {0.f, 0.f, 0.f, 0.f};
  for (int dd = 0; dd < DDIM; ++dd) {
    float w = __shfl(attn, dd, 16);
    const bf16_t* tr = tbf + (long)(rowbase + dd) * HH;
#pragma unroll
    for (int c = 0; c < 4; ++c) acc[c] += w * (float)tr[lane + c * 32];
  }
  long o = (long)node * HH;
#pragma unroll
  for (int c = 0; c < 4; ++c) msg[o + lane + c * 32] = acc[c];
}

// x = tanh(gather(node_msg + node_enc, adj)) * valid   -> bf16
__global__ void xbuild_kernel(const float* __restrict__ msg, const float* __restrict__ enc,
                              const int* __restrict__ adj, const int* __restrict__ nn,
                              bf16_t* __restrict__ xbf) {
  int t = blockIdx.x * blockDim.x + threadIdx.x;  // RR * 32
  int row = t >> 5, c0 = (t & 31) * 4;
  int b = row / (VV * DDIM);
  int a = adj[row];
  float val = (a == nn[b]) ? 0.f : 1.f;
  int ac = a < VV ? (a < 0 ? 0 : a) : VV - 1;
  long src = ((long)b * VV + ac) * HH + c0;
  long o = (long)row * HH + c0;
#pragma unroll
  for (int c = 0; c < 4; ++c) {
    float s = val * (msg[src + c] + enc[src + c]);
    xbf[o + c] = (bf16_t)tanhf(s);
  }
}

// dec_in = [h | h + h[opp]]  -> bf16 (K = 256)
__global__ void decin_kernel(const float* __restrict__ hf, const int* __restrict__ opp,
                             bf16_t* __restrict__ decin) {
  int t = blockIdx.x * blockDim.x + threadIdx.x;  // RR * HH
  int row = t >> 7, c = t & 127;
  int b = row / (VV * DDIM);
  int on = opp[row * 2], od = opp[row * 2 + 1];
  long orow = ((long)b * VV + on) * DDIM + od;
  float h0 = hf[(long)row * HH + c];
  float cm = h0 + hf[orow * HH + c];
  decin[(long)row * 256 + c]       = (bf16_t)h0;
  decin[(long)row * 256 + 128 + c] = (bf16_t)cm;
}

// destination attention: softmax over incoming edges (one wave per node)
__global__ void destattn_kernel(const float* __restrict__ nw, const float* __restrict__ bdec2,
                                const int* __restrict__ in_idx, const int* __restrict__ inv_adj,
                                const int* __restrict__ nn, float* __restrict__ dattn) {
  int wave = threadIdx.x >> 5, lane = threadIdx.x & 31;
  int node = blockIdx.x * 8 + wave;
  int b = node / VV;
  int d = lane & 15;
  int e = node * DDIM + d;
  int iv = in_idx[e * 2], sd = in_idx[e * 2 + 1];
  float v = nw[((long)b * VV + iv) * DDIM + sd] + bdec2[0];
  float m = (inv_adj[e] == nn[b]) ? 1.f : 0.f;
  v -= BIGF * m;
  float mx = v;
#pragma unroll
  for (int o = 8; o; o >>= 1) mx = fmaxf(mx, __shfl_xor(mx, o, 16));
  float ex = __expf(v - mx);
  float s = ex;
#pragma unroll
  for (int o = 8; o; o >>= 1) s += __shfl_xor(s, o, 16);
  if (lane < 16) dattn[e] = ex / s;
}

// regather via rev_indices, mask, then sparsemax over the 16 slots (one thread/node)
__global__ void sparsemax_kernel(const float* __restrict__ dattn, const int* __restrict__ rev,
                                 const int* __restrict__ adj, const int* __restrict__ nn,
                                 float* __restrict__ nrm) {
  int n = blockIdx.x * blockDim.x + threadIdx.x;
  if (n >= NBV) return;
  int b = n / VV, nnb = nn[b];
  float z[DDIM], zs[DDIM];
#pragma unroll
  for (int d = 0; d < DDIM; ++d) {
    int e = n * DDIM + d;
    int rn = rev[e * 2], rd = rev[e * 2 + 1];
    float v = dattn[((long)b * VV + rn) * DDIM + rd];
    float m = (adj[e] == nnb) ? 1.f : 0.f;
    v -= BIGF * m;
    z[d] = v; zs[d] = v;
  }
  for (int i = 1; i < DDIM; ++i) {  // insertion sort, descending
    float key = zs[i]; int j = i - 1;
    while (j >= 0 && zs[j] < key) { zs[j + 1] = zs[j]; --j; }
    zs[j + 1] = key;
  }
  float cum = 0.f, cumk = 0.f; int kz = 0;
#pragma unroll
  for (int k = 0; k < DDIM; ++k) {
    cum += zs[k];
    if (1.f + (float)(k + 1) * zs[k] > cum) { kz = k + 1; cumk = cum; }
  }
  float tau = (cumk - 1.f) / (float)kz;
#pragma unroll
  for (int d = 0; d < DDIM; ++d) {
    int e = n * DDIM + d;
    float valid = (adj[e] == nnb) ? 0.f : 1.f;
    nrm[e] = fmaxf(z[d] - tau, 0.f) * valid;
  }
}

// one proportional min-cost-flow iteration (thread per node, ping-pong buffers)
__global__ void flow_kernel(const float* __restrict__ src, const float* __restrict__ nrm,
                            const float* __restrict__ dem, const int* __restrict__ in_idx,
                            float* __restrict__ dst) {
  int n = blockIdx.x * blockDim.x + threadIdx.x;
  if (n >= NBV) return;
  int b = n / VV;
  float inflow = 0.f;
#pragma unroll
  for (int d = 0; d < DDIM; ++d) {
    int e = n * DDIM + d;
    int iv = in_idx[e * 2], sd = in_idx[e * 2 + 1];
    inflow += src[((long)b * VV + iv) * DDIM + sd];
  }
  float r = fmaxf(inflow - dem[n], 0.f);
#pragma unroll
  for (int d = 0; d < DDIM; ++d) dst[n * DDIM + d] = nrm[n * DDIM + d] * r;
}

__global__ void flowcost_kernel(const float* __restrict__ flow, float* __restrict__ acc) {
  int n = blockIdx.x * blockDim.x + threadIdx.x;
  if (n >= NBV) return;
  int b = n / VV;
  float s = 0.f;
#pragma unroll
  for (int d = 0; d < DDIM; ++d) { float f = flow[n * DDIM + d]; s += f * f; }
  atomicAdd(&acc[b * 4 + 0], s);
}

// node_states = sum_d h  -> bf16
__global__ void nstates_kernel(const float* __restrict__ hf, bf16_t* __restrict__ nsbf) {
  int t = blockIdx.x * blockDim.x + threadIdx.x;  // NBV * HH
  int n = t >> 7, c = t & 127;
  float s = 0.f;
#pragma unroll
  for (int d = 0; d < DDIM; ++d) s += hf[((long)n * DDIM + d) * HH + c];
  nsbf[(long)n * HH + c] = (bf16_t)s;
}

// dual_vars = relu(acc + b_dual2); also accumulate dual_demand per batch
__global__ void dvars_kernel(float* __restrict__ dvars, const float* __restrict__ bdual2,
                             const float* __restrict__ dem, float* __restrict__ acc) {
  int n = blockIdx.x * blockDim.x + threadIdx.x;
  if (n >= NBV) return;
  float v = fmaxf(dvars[n] + bdual2[0], 0.f);
  dvars[n] = v;
  atomicAdd(&acc[(n / VV) * 4 + 2], v * dem[n]);
}

// 10 projected-momentum dual iterations, fully in registers; wave-reduced
// dual_cost contribution (wave spans a single batch b here).
__global__ void dual_kernel(const float* __restrict__ dvars, const int* __restrict__ adj,
                            const int* __restrict__ nn, float* __restrict__ acc) {
  int row = blockIdx.x * blockDim.x + threadIdx.x;  // RR
  int b = row >> 14;          // V*D = 16384
  int node = row >> 4;
  float dv = dvars[node];
  int a = adj[row];
  float valid = (a == nn[b]) ? 0.f : 1.f;
  float mask = 1.f - valid;
  int ac = a < VV ? (a < 0 ? 0 : a) : VV - 1;
  float diff = valid * dvars[b * VV + ac] - mask * dv;
  float f = 0.f, vel = 0.f;
#pragma unroll
  for (int it = 0; it < 10; ++it) {
    float g = 2.f * f + diff;
    vel = 0.9f * vel - 0.01f * g;
    f = fmaxf(f + vel, 0.f);
  }
  f *= valid;
  float contrib = f * f + diff * f;
#pragma unroll
  for (int o = 1; o < 32; o <<= 1) contrib += __shfl_xor(contrib, o, 32);
  if ((threadIdx.x & 31) == 0) atomicAdd(&acc[b * 4 + 1], contrib);
}

__global__ void final_kernel(const float* __restrict__ acc, float* __restrict__ out) {
  int b = threadIdx.x;
  if (b < BB) out[b] = acc[b * 4 + 0] - (acc[b * 4 + 1] - acc[b * 4 + 2]);
}

// ---------------------------------------------------------------------------
// Launcher
// ---------------------------------------------------------------------------
extern "C" void kernel_launch(void* const* d_in, const int* in_sizes, int n_in,
                              void* d_out, int out_size, void* d_ws, size_t ws_size,
                              hipStream_t stream) {
  (void)in_sizes; (void)n_in; (void)out_size; (void)ws_size;

  const float* demands  = (const float*)d_in[0];
  const float* feat     = (const float*)d_in[1];
  const float* node_emb = (const float*)d_in[4];
  const float* W_enc    = (const float*)d_in[5];
  const float* b_enc    = (const float*)d_in[6];
  const float* W_gat    = (const float*)d_in[7];
  const float* b_gat    = (const float*)d_in[8];
  const float* a_gat    = (const float*)d_in[9];
  const float* W_gru_x  = (const float*)d_in[10];
  const float* W_gru_h  = (const float*)d_in[11];
  const float* b_gru    = (const float*)d_in[12];
  const float* W_dec1   = (const float*)d_in[13];
  const float* b_dec1   = (const float*)d_in[14];
  const float* W_dec2   = (const float*)d_in[15];
  const float* b_dec2   = (const float*)d_in[16];
  const float* W_dual1  = (const float*)d_in[17];
  const float* b_dual1  = (const float*)d_in[18];
  const float* W_dual2  = (const float*)d_in[19];
  const float* b_dual2  = (const float*)d_in[20];
  const int* adj     = (const int*)d_in[21];
  const int* inv_adj = (const int*)d_in[22];
  const int* in_idx  = (const int*)d_in[23];
  const int* rev_idx = (const int*)d_in[24];
  const int* opp_idx = (const int*)d_in[25];
  const int* nn      = (const int*)d_in[26];
  float* out = (float*)d_out;

  char* ws = (char*)d_ws;
  size_t off = 0;
  auto alloc = [&](size_t bytes) -> void* {
    off = (off + 255) & ~(size_t)255;
    void* p = ws + off;
    off += bytes;
    return p;
  };

  bf16_t* WENC   = (bf16_t*)alloc((size_t)HH * KENC * 2);
  bf16_t* WGAT   = (bf16_t*)alloc((size_t)HH * HH * 2);
  bf16_t* WGRUX  = (bf16_t*)alloc((size_t)3 * HH * HH * 2);
  bf16_t* WGRUH  = (bf16_t*)alloc((size_t)3 * HH * HH * 2);
  bf16_t* WDEC1  = (bf16_t*)alloc((size_t)HDD * 256 * 2);
  bf16_t* WDUAL1 = (bf16_t*)alloc((size_t)HDD * HH * 2);
  bf16_t* EMBN   = (bf16_t*)alloc((size_t)VV * EE * 2);
  bf16_t* AENC   = (bf16_t*)alloc((size_t)NBV * KENC * 2);
  float*  NODEENC = (float*)alloc((size_t)NBV * HH * 4);
  float*  NODEMSG = (float*)alloc((size_t)NBV * HH * 4);
  bf16_t* NSBF    = (bf16_t*)alloc((size_t)NBV * HH * 2);
  float*  HF      = (float*)alloc((size_t)RR * HH * 4);
  bf16_t* HBF     = (bf16_t*)alloc((size_t)RR * HH * 2);
  bf16_t* TBF     = (bf16_t*)alloc((size_t)RR * HH * 2);   // also DECIN rows [0, RR/2)
  bf16_t* XBF     = (bf16_t*)alloc((size_t)RR * HH * 2);   // also DECIN rows [RR/2, RR)
  float*  LOGITS  = (float*)alloc((size_t)RR * 4);
  float*  NW      = (float*)alloc((size_t)RR * 4);
  float*  DATTN   = (float*)alloc((size_t)RR * 4);
  float*  NORM    = (float*)alloc((size_t)RR * 4);
  float*  FLOWA   = (float*)alloc((size_t)RR * 4);
  float*  FLOWB   = (float*)alloc((size_t)RR * 4);
  float*  DVARS   = (float*)alloc((size_t)NBV * 4);
  float*  ACC     = (float*)alloc((size_t)BB * 4 * 4);
  bf16_t* DECIN   = TBF;  // TBF and XBF are contiguous: RR x 256 bf16

  const int T = 256;

  // --- pack weights (f32 [K,N] -> bf16 [n*Kp + k]) ---
  packw_kernel<<<(HH * KENC + T - 1) / T, T, 0, stream>>>(W_enc, WENC, 80, KENC, HH);
  packw_kernel<<<(HH * HH + T - 1) / T, T, 0, stream>>>(W_gat, WGAT, HH, HH, HH);
  packw_kernel<<<(3 * HH * HH + T - 1) / T, T, 0, stream>>>(W_gru_x, WGRUX, HH, HH, 3 * HH);
  packw_kernel<<<(3 * HH * HH + T - 1) / T, T, 0, stream>>>(W_gru_h, WGRUH, HH, HH, 3 * HH);
  packw_kernel<<<(HDD * 256 + T - 1) / T, T, 0, stream>>>(W_dec1, WDEC1, 256, 256, HDD);
  packw_kernel<<<(HDD * HH + T - 1) / T, T, 0, stream>>>(W_dual1, WDUAL1, HH, HH, HDD);

  // --- encoder ---
  embn_kernel<<<(VV + T - 1) / T, T, 0, stream>>>(node_emb, EMBN);
  aenc_kernel<<<(NBV * KENC) / T, T, 0, stream>>>(EMBN, feat, AENC);
  gemm_bf16_kernel<<<dim3(NBV / 128, HH / 64), T, 0, stream>>>(
      AENC, WENC, b_enc, NODEENC, nullptr, nullptr, nullptr, HH, KENC, 0);
  hinit_kernel<<<(RR * 32) / T, T, 0, stream>>>(NODEENC, adj, nn, HF, HBF);

  // --- 2 DirectionalGAT + GRU layers (bf16 h ping-pongs HBF <-> TBF) ---
  bf16_t* hb = HBF;
  bf16_t* tb = TBF;
  for (int layer = 0; layer < 2; ++layer) {
    zero_kernel<<<(RR + T - 1) / T, T, 0, stream>>>(LOGITS, RR);
    gemm_bf16_kernel<<<dim3(RR / 128, HH / 64), T, 0, stream>>>(
        hb, WGAT, b_gat, nullptr, tb, a_gat, LOGITS, HH, HH, 1);
    attn_kernel<<<NBV / 8, T, 0, stream>>>(LOGITS, tb, adj, nn, NODEMSG);
    xbuild_kernel<<<(RR * 32) / T, T, 0, stream>>>(NODEMSG, NODEENC, adj, nn, XBF);
    gru_kernel<<<RR / 128, T, 0, stream>>>(XBF, hb, WGRUX, WGRUH, b_gru, HF, tb);
    bf16_t* tmp = hb; hb = tb; tb = tmp;  // new bf16 h is in tb's old buffer
  }
  // after 2 layers: f32 h in HF; bf16 h in HBF; TBF/XBF free -> DECIN region

  // --- decoder ---
  decin_kernel<<<(RR * HH) / T, T, 0, stream>>>(HF, opp_idx, DECIN);
  zero_kernel<<<(RR + T - 1) / T, T, 0, stream>>>(NW, RR);
  gemm_bf16_kernel<<<dim3(RR / 128, 1), T, 0, stream>>>(
      DECIN, WDEC1, b_dec1, nullptr, nullptr, W_dec2, NW, HDD, 256, 1);
  destattn_kernel<<<NBV / 8, T, 0, stream>>>(NW, b_dec2, in_idx, inv_adj, nn, DATTN);
  sparsemax_kernel<<<NBV / T, T, 0, stream>>>(DATTN, rev_idx, adj, nn, NORM);

  // --- min-cost-flow iterations (ping-pong) ---
  zero_kernel<<<(RR + T - 1) / T, T, 0, stream>>>(FLOWA, RR);
  zero_kernel<<<1, T, 0, stream>>>(ACC, BB * 4);
  for (int it = 0; it < 10; ++it) {
    const float* s = (it & 1) ? FLOWB : FLOWA;
    float* dst     = (it & 1) ? FLOWA : FLOWB;
    flow_kernel<<<NBV / T, T, 0, stream>>>(s, NORM, demands, in_idx, dst);
  }
  flowcost_kernel<<<NBV / T, T, 0, stream>>>(FLOWA, ACC);

  // --- dual variables + dual flow ---
  nstates_kernel<<<(NBV * HH) / T, T, 0, stream>>>(HF, NSBF);
  zero_kernel<<<(NBV + T - 1) / T, T, 0, stream>>>(DVARS, NBV);
  gemm_bf16_kernel<<<dim3(NBV / 128, 1), T, 0, stream>>>(
      NSBF, WDUAL1, b_dual1, nullptr, nullptr, W_dual2, DVARS, HDD, HH, 1);
  dvars_kernel<<<NBV / T, T, 0, stream>>>(DVARS, b_dual2, demands, ACC);
  dual_kernel<<<RR / T, T, 0, stream>>>(DVARS, adj, nn, ACC);

  final_kernel<<<1, 32, 0, stream>>>(ACC, out);
}
